// TemporalDAGConstraint_13108240187549
// MI455X (gfx1250) — matrix operand order repaired and checked by reading, not compiled, not run on hardware
//
#include <hip/hip_runtime.h>
#include <hip/hip_bf16.h>
#include <stdint.h>

// ---------------------------------------------------------------------------
// h = trace(expm(A0 * A0)) - n  via trace-Taylor:
//   B = A0 .* A0  (bf16),  h = sum_{k=1..KMAX} tr(B^k)/k!
// B^k computed by a chain of bf16 WMMA GEMMs with fp32 accumulation.
// ---------------------------------------------------------------------------

typedef __bf16 bf16_t;
typedef __bf16 v16bf __attribute__((ext_vector_type(16)));
typedef __bf16 v8bf  __attribute__((ext_vector_type(8)));
typedef float  v8f   __attribute__((ext_vector_type(8)));
typedef int    v4i   __attribute__((ext_vector_type(4)));

#define NN   2048
#define KMAX 14          // highest matrix power in the Taylor series

static constexpr int BM  = 128;   // block tile M
static constexpr int BN  = 128;   // block tile N
static constexpr int BK  = 32;    // K per WMMA / per LDS stage
static constexpr int LDT = 56;    // LDS row stride (bf16): 112B, 16B aligned, conflict-light

// ---- optional CDNA5 async global->LDS path (probe-guarded) ----------------
#if defined(__has_builtin)
#if __has_builtin(__builtin_amdgcn_global_load_async_to_lds_b128) && \
    __has_builtin(__builtin_amdgcn_s_wait_asynccnt)
#define USE_ASYNC_LDS 1
#endif
#endif
#ifndef USE_ASYNC_LDS
#define USE_ASYNC_LDS 0
#endif

#if USE_ASYNC_LDS
typedef __attribute__((address_space(1))) v4i* as1_v4i;   // global int4*
typedef __attribute__((address_space(3))) v4i* as3_v4i;   // LDS int4*
#endif

// -------------------------------- init ------------------------------------
__global__ void zero_tr_kernel(float* tr) {
    if (threadIdx.x < 32) tr[threadIdx.x] = 0.0f;
}

__global__ __launch_bounds__(256)
void square_to_bf16_kernel(const float* __restrict__ A0,
                           bf16_t* __restrict__ Bb,
                           float* __restrict__ tr) {
    int i = blockIdx.x * 256 + threadIdx.x;          // i < 2048*2048
    float a = A0[i];
    float b = a * a;
    Bb[i] = (bf16_t)b;
    int r = i >> 11;
    int c = i & (NN - 1);
    if (r == c) atomicAdd(&tr[1], b);                // tr(B) exactly in fp32
}

// ------------------------------- GEMM -------------------------------------
// Pn = P * B  (all 2048x2048, bf16 in / bf16 out, fp32 accumulate),
// diagonal of the fp32 result atomically accumulated into *tracc.
__global__ __launch_bounds__(256)
void gemm_bf16_trace_kernel(const bf16_t* __restrict__ P,
                            const bf16_t* __restrict__ B,
                            bf16_t* __restrict__ Pn,
                            float* __restrict__ tracc) {
    __shared__ bf16_t sA[2][BM * LDT];   // 2 * 128*56*2B = 28 KB
    __shared__ bf16_t sB[2][BN * LDT];   // 2 * 128*56*2B = 28 KB (B transposed: [n][k])

    const int tid   = threadIdx.x;
    const int lane  = tid & 31;
    const int wid   = tid >> 5;          // 8 waves
    const int waveM = wid & 3;           // 4 waves along M -> 4*32 = 128
    const int waveN = wid >> 2;          // 2 waves along N -> 2*64 = 128
    const int bm    = blockIdx.y * BM;
    const int bn    = blockIdx.x * BN;

    // Stage a 128x32 bf16 tile of P, row-major, 16B chunks per lane.
    auto stageA = [&](int k0, bf16_t* dst) {
#pragma unroll
        for (int it = 0; it < 2; ++it) {
            int c   = tid + it * 256;    // 512 chunks of 8 bf16
            int row = c >> 2;
            int co  = (c & 3) * 8;
            const bf16_t* src = P + (size_t)(bm + row) * NN + k0 + co;
            bf16_t* d = dst + row * LDT + co;
#if USE_ASYNC_LDS
            __builtin_amdgcn_global_load_async_to_lds_b128(
                (as1_v4i)(bf16_t*)src, (as3_v4i)d, /*offset=*/0, /*cpol=*/0);
#else
            *(v8bf*)d = *(const v8bf*)src;
#endif
        }
    };
    // Stage a 32(K) x 128(N) tile of B transposed into LDS as [n][k].
    auto stageB = [&](int k0, bf16_t* dst) {
        int n  = tid & 127;
        int kq = (tid >> 7) * 16;
        v8bf v0, v1;
#pragma unroll
        for (int i = 0; i < 8; ++i)
            v0[i] = B[(size_t)(k0 + kq + i) * NN + bn + n];
#pragma unroll
        for (int i = 0; i < 8; ++i)
            v1[i] = B[(size_t)(k0 + kq + 8 + i) * NN + bn + n];
        *(v8bf*)(dst + n * LDT + kq)     = v0;
        *(v8bf*)(dst + n * LDT + kq + 8) = v1;
    };

    v8f acc[2][4] = {};

    stageA(0, sA[0]);
    stageB(0, sB[0]);
#if USE_ASYNC_LDS
    __builtin_amdgcn_s_wait_asynccnt(0);
#endif
    __syncthreads();

    const int mrow = waveM * 32 + (lane & 15);
    const int ncol = waveN * 64 + (lane & 15);
    const int kbA  = (lane >> 4) * 8;    // A operand: lanes<16 hold K in [0,8)+[16,24)
    const int kbB  = (lane >> 4) * 8;    // B operand: lanes<16 hold K in [0,8)+[16,24)

    for (int kt = 0; kt < NN / BK; ++kt) {
        const int cur = kt & 1;
        if (kt + 1 < NN / BK) {          // prefetch next K-slab into other buffer
            stageA((kt + 1) * BK, sA[cur ^ 1]);
            stageB((kt + 1) * BK, sB[cur ^ 1]);
        }

        const bf16_t* a0 = sA[cur];
        const bf16_t* b0 = sB[cur];

        v16bf afrag[2], bfrag[4];
#pragma unroll
        for (int mt = 0; mt < 2; ++mt) {
            const bf16_t* pa = a0 + (mrow + mt * 16) * LDT;
            v8bf lo = *(const v8bf*)(pa + kbA);
            v8bf hi = *(const v8bf*)(pa + kbA + 16);
            afrag[mt] = __builtin_shufflevector(lo, hi, 0,1,2,3,4,5,6,7,8,9,10,11,12,13,14,15);
        }
#pragma unroll
        for (int nt = 0; nt < 4; ++nt) {
            const bf16_t* pb = b0 + (ncol + nt * 16) * LDT + kbB;
            v8bf lo = *(const v8bf*)(pb);
            v8bf hi = *(const v8bf*)(pb + 16);
            bfrag[nt] = __builtin_shufflevector(lo, hi, 0,1,2,3,4,5,6,7,8,9,10,11,12,13,14,15);
        }

#pragma unroll
        for (int mt = 0; mt < 2; ++mt)
#pragma unroll
            for (int nt = 0; nt < 4; ++nt)
                acc[mt][nt] = __builtin_amdgcn_wmma_f32_16x16x32_bf16(
                    /*neg_a=*/false, afrag[mt],
                    /*neg_b=*/false, bfrag[nt],
                    /*c_mod=*/(short)0, acc[mt][nt],
                    /*reuse_a=*/false, /*reuse_b=*/false);

#if USE_ASYNC_LDS
        __builtin_amdgcn_s_wait_asynccnt(0);   // copies for slab kt+1 done
#endif
        __syncthreads();
    }

    // Epilogue: bf16 store for the next power + diagonal harvest (fp32).
#pragma unroll
    for (int mt = 0; mt < 2; ++mt) {
#pragma unroll
        for (int nt = 0; nt < 4; ++nt) {
            const int gcol = bn + waveN * 64 + nt * 16 + (lane & 15);
            const int gr0  = bm + waveM * 32 + mt * 16 + 8 * (lane >> 4);
#pragma unroll
            for (int j = 0; j < 8; ++j) {
                const int gr = gr0 + j;
                const float v = acc[mt][nt][j];
                Pn[(size_t)gr * NN + gcol] = (bf16_t)v;
                if (gr == gcol) atomicAdd(tracc, v);
            }
        }
    }
}

// ------------------------------ finalize ----------------------------------
__global__ void finalize_kernel(const float* __restrict__ tr, float* __restrict__ out) {
    const float invf[KMAX + 1] = {
        0.0f,
        1.0f,
        1.0f / 2.0f,
        1.0f / 6.0f,
        1.0f / 24.0f,
        1.0f / 120.0f,
        1.0f / 720.0f,
        1.0f / 5040.0f,
        1.0f / 40320.0f,
        1.0f / 362880.0f,
        1.0f / 3628800.0f,
        1.0f / 39916800.0f,
        1.0f / 479001600.0f,
        1.0f / 6227020800.0f,
        1.0f / 87178291200.0f
    };
    float h = 0.0f;
    for (int k = 1; k <= KMAX; ++k) h += tr[k] * invf[k];
    out[0] = h;   // trace(expm(B)) - n
}

// ------------------------------ launcher ----------------------------------
extern "C" void kernel_launch(void* const* d_in, const int* in_sizes, int n_in,
                              void* d_out, int out_size, void* d_ws, size_t ws_size,
                              hipStream_t stream) {
    (void)in_sizes; (void)n_in; (void)out_size; (void)ws_size;

    const float* A = (const float*)d_in[0];        // slice 0 = first 2048*2048 floats
    float* out = (float*)d_out;

    uint8_t* ws = (uint8_t*)d_ws;
    bf16_t* Bb = (bf16_t*)(ws);                      //  8 MiB : B   (bf16)
    bf16_t* P0 = (bf16_t*)(ws + (size_t(8)  << 20)); //  8 MiB : ping
    bf16_t* P1 = (bf16_t*)(ws + (size_t(16) << 20)); //  8 MiB : pong
    float*  tr = (float*)(ws + (size_t(24) << 20));  //  trace slots [0..31]

    zero_tr_kernel<<<1, 32, 0, stream>>>(tr);
    square_to_bf16_kernel<<<(NN * NN) / 256, 256, 0, stream>>>(A, Bb, tr);

    dim3 grid(NN / BN, NN / BM);   // (16, 16)
    const bf16_t* src = Bb;
    bf16_t* dsts[2] = {P0, P1};
    for (int k = 2; k <= KMAX; ++k) {
        bf16_t* dst = dsts[k & 1];
        gemm_bf16_trace_kernel<<<grid, 256, 0, stream>>>(src, Bb, dst, tr + k);
        src = dst;
    }

    finalize_kernel<<<1, 1, 0, stream>>>(tr, out);
}
